// Transformer_43954695307574
// MI455X (gfx1250) — matrix operand order repaired and checked by reading, not compiled, not run on hardware
//
#include <hip/hip_runtime.h>
#include <hip/hip_bf16.h>
#include <math.h>
#include <stdint.h>

// ---------------------------------------------------------------------------
// CDNA5 (gfx1250, wave32) transformer forward, v3.
//  * f16 WMMA GEMM, BM=128 x BN=64 x BK=64: 8 waves, 4 acc/wave,
//    8 v_wmma_f32_16x16x32_f16 per K-tile per wave (4x A-frag reuse)
//  * double-buffered LDS tiles filled with GLOBAL_LOAD_ASYNC_TO_LDS_B128;
//    software pipeline overlaps DMA of tile kt+1 with WMMA on tile kt
//    (s_wait_asynccnt 6 = previous tile's 6 in-order ops complete)
//  * weights transpose-converted to f16 [N,K] once -> every GEMM is
//    A[M,K] @ Bt[N,K]^T with 16B-contiguous chunks on both operands
// ---------------------------------------------------------------------------

typedef __attribute__((ext_vector_type(16))) _Float16 v16h;
typedef __attribute__((ext_vector_type(8)))  _Float16 h8;
typedef __attribute__((ext_vector_type(8)))  float    v8f;

#define TB  256   // 8 waves
#define BM  128
#define BN  64
#define BK  64
#define LDK (BK + 8)   // 144B row stride, 16B-aligned chunks

#define USE_ASYNC_LDS 1

union Frag16 { v16h v; h8 h[2]; };

// 16-byte global -> LDS async DMA (ASYNCcnt). vdst = per-lane LDS byte addr
// (generic LDS pointers truncate to the LDS offset in the low 32 bits).
__device__ __forceinline__ void copy16_g2l(const _Float16* g, _Float16* l)
{
#if USE_ASYNC_LDS
    unsigned lds = (unsigned)(uintptr_t)l;
    asm volatile("global_load_async_to_lds_b128 %0, %1, off"
                 :: "v"(lds), "v"(g) : "memory");
#else
    *(h8*)l = *(const h8*)g;
#endif
}
__device__ __forceinline__ void async_wait0()
{
#if USE_ASYNC_LDS
    asm volatile("s_wait_asynccnt 0x0" ::: "memory");
#endif
}
__device__ __forceinline__ void async_wait6()   // prev tile's 6 ops retired
{
#if USE_ASYNC_LDS
    asm volatile("s_wait_asynccnt 0x6" ::: "memory");
#endif
}

// ---------------------------------------------------------------------------
// Universal batched GEMM:  out = alpha * (A @ Bt^T) + bias  [+ ReLU]
//   A : f16 [M,K] (lda)      Bt : f16 [N,K] (ldb)
//   C : f32 (optional)       Ch : f16 mirror (optional)
//   vtmode: Ch written as per-head transposed V: ((b*vtH+h)*vtDK+d)*vtS + s
// Requires M%128==0, N%64==0, K%64==0 (holds for every GEMM here).
__global__ __launch_bounds__(TB)
void gemm_f16_wmma(const _Float16* __restrict__ A, const _Float16* __restrict__ Bt,
                   const float* __restrict__ bias,
                   float* __restrict__ C, _Float16* __restrict__ Ch,
                   int M, int N, int K, int lda, int ldb, int ldc, int ldh,
                   long long sAo, long long sAi, long long sBo, long long sBi,
                   long long sCo, long long sCi, long long sHo, long long sHi,
                   int Hdiv, float alpha, int relu,
                   int vtmode, int vtS, int vtH, int vtDK)
{
    __shared__ _Float16 As[2][BM][LDK];   // 2 x 18KB
    __shared__ _Float16 Bs[2][BN][LDK];   // 2 x  9KB

    const int zo = blockIdx.z / Hdiv;
    const int zi = blockIdx.z % Hdiv;
    A  += (size_t)zo * sAo + (size_t)zi * sAi;
    Bt += (size_t)zo * sBo + (size_t)zi * sBi;
    if (C)  C  += (size_t)zo * sCo + (size_t)zi * sCi;
    if (Ch) Ch += (size_t)zo * sHo + (size_t)zi * sHi;

    const int n0 = blockIdx.x * BN;
    const int m0 = blockIdx.y * BM;

    const int tid  = threadIdx.x;
    const int wave = tid >> 5;         // 0..7 -> 16-row strip
    const int lane = tid & 31;
    const int half = lane >> 4;
    const int l16  = lane & 15;

    // stage one 128x64 A tile + 64x64 B tile into buffer bi (6 ops/thread)
    auto stage = [&](int bi, int k0) {
        #pragma unroll
        for (int c = 0; c < 4; ++c) {           // A: 1024 chunks
            const int cc  = tid + c * TB;
            const int row = cc >> 3;
            const int col = (cc & 7) * 8;
            copy16_g2l(A + (size_t)(m0 + row) * lda + k0 + col, &As[bi][row][col]);
        }
        #pragma unroll
        for (int c = 0; c < 2; ++c) {           // B: 512 chunks
            const int cc  = tid + c * TB;
            const int row = cc >> 3;
            const int col = (cc & 7) * 8;
            copy16_g2l(Bt + (size_t)(n0 + row) * ldb + k0 + col, &Bs[bi][row][col]);
        }
    };

    v8f acc[4] = {v8f{}, v8f{}, v8f{}, v8f{}};

    const int kTiles = K / BK;
    stage(0, 0);                                 // prologue: tile 0 -> buf 0
    for (int kt = 0; kt < kTiles; ++kt) {
        const int cur = kt & 1;
        if (kt + 1 < kTiles) {
            stage(cur ^ 1, (kt + 1) * BK);       // DMA next tile while computing
            __builtin_prefetch(A + (size_t)(m0 + (tid >> 3)) * lda + (kt + 1) * BK, 0, 3);
            async_wait6();                       // tile kt's 6 ops retired
        } else {
            async_wait0();
        }
        __syncthreads();

        #pragma unroll
        for (int ks = 0; ks < 2; ++ks) {
            const int kb = ks * 32;
            // A frag 16x32: lane=M, K chunks at kb+half*8 / +16
            Frag16 a;
            const int mrow = wave * 16 + l16;
            a.h[0] = *(const h8*)&As[cur][mrow][kb + half * 8];
            a.h[1] = *(const h8*)&As[cur][mrow][kb + half * 8 + 16];
            #pragma unroll
            for (int sn = 0; sn < 4; ++sn) {
                // B frag 32x16: lane=N, K chunks at kb+half*8 / +16 (same as A)
                Frag16 b;
                const int ncol = sn * 16 + l16;
                b.h[0] = *(const h8*)&Bs[cur][ncol][kb + half * 8];
                b.h[1] = *(const h8*)&Bs[cur][ncol][kb + half * 8 + 16];
                acc[sn] = __builtin_amdgcn_wmma_f32_16x16x32_f16(
                    false, a.v, false, b.v, (short)0, acc[sn], false, false);
            }
        }
        __syncthreads();   // all waves done reading buf cur before overwrite
    }

    // epilogue: D layout -> row = wave*16 + half*8 + i, col = sn*16 + l16
    const int rbase = m0 + wave * 16 + half * 8;
    #pragma unroll
    for (int sn = 0; sn < 4; ++sn) {
        const int col = n0 + sn * 16 + l16;
        const float bv = bias ? bias[col] : 0.0f;
        float vals[8];
        #pragma unroll
        for (int i = 0; i < 8; ++i) {
            float vo = acc[sn][i] * alpha + bv;
            vals[i] = relu ? (vo > 0.0f ? vo : 0.0f) : vo;
        }
        if (C) {
            #pragma unroll
            for (int i = 0; i < 8; ++i)
                C[(size_t)(rbase + i) * ldc + col] = vals[i];
        }
        if (Ch) {
            if (!vtmode) {
                #pragma unroll
                for (int i = 0; i < 8; ++i)
                    Ch[(size_t)(rbase + i) * ldh + col] = (_Float16)vals[i];
            } else {
                // V-transpose mirror: rows rbase..rbase+7 are contiguous s
                const int b = rbase / vtS, s = rbase % vtS;
                const int h = col / vtDK,  d = col % vtDK;
                const size_t base = (((size_t)b * vtH + h) * vtDK + d) * vtS + s;
                h8 p;
                #pragma unroll
                for (int i = 0; i < 8; ++i) p[i] = (_Float16)vals[i];
                *(h8*)&Ch[base] = p;
            }
        }
    }
}

// ---------------------------------------------------------------------------
// f32 [K,N] -> f16 [N,K] transpose-convert (z = layer, stride K*N)
__global__ __launch_bounds__(TB)
void transpose_cvt(const float* __restrict__ in, _Float16* __restrict__ out,
                   int K, int N)
{
    __shared__ float tile[32][33];
    const size_t zoff = (size_t)blockIdx.z * K * N;
    in += zoff; out += zoff;
    const int n0 = blockIdx.x * 32, k0 = blockIdx.y * 32;
    const int tx = threadIdx.x & 31, ty = threadIdx.x >> 5;
    #pragma unroll
    for (int j = 0; j < 4; ++j)
        tile[ty + j * 8][tx] = in[(size_t)(k0 + ty + j * 8) * N + n0 + tx];
    __syncthreads();
    #pragma unroll
    for (int j = 0; j < 4; ++j)
        out[(size_t)(n0 + ty + j * 8) * K + k0 + tx] = (_Float16)tile[tx][ty + j * 8];
}

// elementwise f32 -> f16
__global__ __launch_bounds__(TB)
void cvt_f32_f16(const float* __restrict__ in, _Float16* __restrict__ out, size_t n)
{
    for (size_t i = (size_t)blockIdx.x * TB + threadIdx.x; i < n;
         i += (size_t)gridDim.x * TB)
        out[i] = (_Float16)in[i];
}

// ---------------------------------------------------------------------------
// Embedding + sinusoidal PE -> f32 x and f16 mirror
__global__ __launch_bounds__(TB)
void embed_pe(const int* __restrict__ tok, const float* __restrict__ emb,
              float* __restrict__ out, _Float16* __restrict__ outh, int S, int D)
{
    const int t   = blockIdx.x;
    const int pos = t % S;
    const int id  = tok[t];
    const float ef = sqrtf((float)D);
    const float ln10k = logf(10000.0f);
    for (int c = threadIdx.x; c < D; c += TB) {
        const int i = c >> 1;
        const float freq = __expf(-(2.0f * (float)i / (float)D) * ln10k);
        const float ang  = (float)pos * freq;
        const float pe   = (c & 1) ? cosf(ang) : sinf(ang);
        const float vo   = emb[(size_t)id * D + c] * ef + pe;
        out[(size_t)t * D + c]  = vo;
        outh[(size_t)t * D + c] = (_Float16)vo;
    }
}

// ---------------------------------------------------------------------------
// Row softmax with causal/key-padding masks; writes f32 in-place + f16 mirror
__global__ __launch_bounds__(TB)
void softmax_mask(float* __restrict__ sc, _Float16* __restrict__ sch,
                  const int* __restrict__ pad, int S, int H, int causal)
{
    __shared__ float red[TB];
    const int q  = blockIdx.x;
    const int bh = blockIdx.y;
    const int b  = bh / H;
    const size_t ro = ((size_t)bh * S + q) * S;
    float* row = sc + ro;
    _Float16* rowh = sch + ro;
    const int tid = threadIdx.x;

    float v0 = row[tid], v1 = row[tid + TB];
    if (causal) {
        if (tid      > q) v0 = -INFINITY;
        if (tid + TB > q) v1 = -INFINITY;
    }
    if (pad) {
        if (pad[(size_t)b * S + tid]      == 0) v0 = -INFINITY;
        if (pad[(size_t)b * S + tid + TB] == 0) v1 = -INFINITY;
    }
    red[tid] = fmaxf(v0, v1); __syncthreads();
    for (int s = TB / 2; s > 0; s >>= 1) {
        if (tid < s) red[tid] = fmaxf(red[tid], red[tid + s]);
        __syncthreads();
    }
    const float m = red[0]; __syncthreads();

    const float e0 = __expf(v0 - m), e1 = __expf(v1 - m);
    red[tid] = e0 + e1; __syncthreads();
    for (int s = TB / 2; s > 0; s >>= 1) {
        if (tid < s) red[tid] += red[tid + s];
        __syncthreads();
    }
    const float inv = 1.0f / red[0];
    row[tid]       = e0 * inv;
    row[tid + TB]  = e1 * inv;
    rowh[tid]      = (_Float16)(e0 * inv);
    rowh[tid + TB] = (_Float16)(e1 * inv);
}

// ---------------------------------------------------------------------------
// out = LayerNorm(x + delta) * s + b ; writes f32 out and f16 mirror
__global__ __launch_bounds__(TB)
void add_layernorm(const float* __restrict__ x, const float* __restrict__ delta,
                   const float* __restrict__ s, const float* __restrict__ b,
                   float* __restrict__ out, _Float16* __restrict__ outh, int D)
{
    __shared__ float red[TB];
    const int row = blockIdx.x;
    const int tid = threadIdx.x;
    float v[4];
    float sum = 0.0f;
    #pragma unroll
    for (int i = 0; i < 4; ++i) {
        const int c = tid + i * TB;
        v[i] = x[(size_t)row * D + c] + delta[(size_t)row * D + c];
        sum += v[i];
    }
    red[tid] = sum; __syncthreads();
    for (int st = TB / 2; st > 0; st >>= 1) {
        if (tid < st) red[tid] += red[tid + st];
        __syncthreads();
    }
    const float mean = red[0] / (float)D; __syncthreads();

    float sq = 0.0f;
    #pragma unroll
    for (int i = 0; i < 4; ++i) { const float d = v[i] - mean; sq += d * d; }
    red[tid] = sq; __syncthreads();
    for (int st = TB / 2; st > 0; st >>= 1) {
        if (tid < st) red[tid] += red[tid + st];
        __syncthreads();
    }
    const float rstd = rsqrtf(red[0] / (float)D + 1e-5f);
    #pragma unroll
    for (int i = 0; i < 4; ++i) {
        const int c = tid + i * TB;
        const float vo = (v[i] - mean) * rstd * s[c] + b[c];
        out[(size_t)row * D + c]  = vo;
        outh[(size_t)row * D + c] = (_Float16)vo;
    }
}

// ---------------------------------------------------------------------------
extern "C" void kernel_launch(void* const* d_in, const int* in_sizes, int n_in,
                              void* d_out, int out_size, void* d_ws, size_t ws_size,
                              hipStream_t stream)
{
    (void)in_sizes; (void)n_in; (void)out_size; (void)ws_size;

    const int D = 1024, H = 16, DK = 64, F = 4096, V = 32000, B = 4, S = 512;
    const int T = B * S;
    const size_t DD = (size_t)D * D, DF = (size_t)D * F;
    const size_t TD = (size_t)T * D, TF = (size_t)T * F;
    const size_t BHSS = (size_t)B * H * S * S;

    const int*   dec_x = (const int*)d_in[0];
    const int*   enc_x = (const int*)d_in[1];
    const int*   pad   = (const int*)d_in[2];
    const float* emb   = (const float*)d_in[3];
    #define P(i) ((const float*)d_in[(i)])

    // ---------------- workspace ----------------
    float* x  = (float*)d_ws;            // [T,D] f32 stream
    float* t1 = x  + TD;                 // [T,D] f32 pre-LN delta
    float* sc = t1 + TD;                 // [B,H,S,S] f32 scores
    _Float16* hcur = (_Float16*)(sc + BHSS);
    auto takeH = [&](size_t n) { _Float16* p = hcur; hcur += n; return p; };
    _Float16* xh   = takeH(TD);
    _Float16* memh = takeH(TD);
    _Float16* qh   = takeH(TD);
    _Float16* kh   = takeH(TD);
    _Float16* vT   = takeH(TD);          // [B,H,DK,S]
    _Float16* t0h  = takeH(TD);
    _Float16* ffh  = takeH(TF);
    _Float16* sch  = takeH(BHSS);
    // f16 transposed weights [N,K] (per-layer stride = K*N)
    _Float16* embh  = takeH((size_t)V * D);
    _Float16* eW[4]; for (int i = 0; i < 4; ++i) eW[i] = takeH(6 * DD);
    _Float16* eW1   = takeH(6 * DF);
    _Float16* eW2   = takeH(6 * DF);
    _Float16* dW[8]; for (int i = 0; i < 8; ++i) dW[i] = takeH(6 * DD);
    _Float16* dW1   = takeH(6 * DF);
    _Float16* dW2   = takeH(6 * DF);

    // ---------------- weight conversion (once per call) ----------------
    cvt_f32_f16<<<2048, TB, 0, stream>>>(emb, embh, (size_t)V * D);
    for (int i = 0; i < 4; ++i)   // enc wq,wk,wv,wo
        transpose_cvt<<<dim3(D/32, D/32, 6), TB, 0, stream>>>(P(4 + 2*i), eW[i], D, D);
    transpose_cvt<<<dim3(F/32, D/32, 6), TB, 0, stream>>>(P(14), eW1, D, F);
    transpose_cvt<<<dim3(D/32, F/32, 6), TB, 0, stream>>>(P(16), eW2, F, D);
    for (int i = 0; i < 8; ++i)   // dec wq..wo, cwq..cwo
        transpose_cvt<<<dim3(D/32, D/32, 6), TB, 0, stream>>>(P(20 + 2*i), dW[i], D, D);
    transpose_cvt<<<dim3(F/32, D/32, 6), TB, 0, stream>>>(P(42), dW1, D, F);
    transpose_cvt<<<dim3(D/32, F/32, 6), TB, 0, stream>>>(P(44), dW2, F, D);

    // ---------------- GEMM wrapper ----------------
    auto gemm = [&](const _Float16* A, const _Float16* Bt, const float* bias,
                    float* C, _Float16* Ch,
                    int M, int N, int K, int lda, int ldb, int ldc, int ldh,
                    long long sAo, long long sAi, long long sBo, long long sBi,
                    long long sCo, long long sCi, long long sHo, long long sHi,
                    int Hdiv, int nb, float alpha, int relu, int vtmode) {
        dim3 g(N / BN, M / BM, nb);
        gemm_f16_wmma<<<g, TB, 0, stream>>>(A, Bt, bias, C, Ch,
            M, N, K, lda, ldb, ldc, ldh, sAo, sAi, sBo, sBi,
            sCo, sCi, sHo, sHi, Hdiv, alpha, relu, vtmode, S, H, DK);
    };

    const float iskd = 1.0f / sqrtf((float)DK);
    const long long SD = (long long)S * D, SS = (long long)S * S;
    const long long DKS = (long long)DK * S;

    auto mha = [&](const _Float16* xqh, const _Float16* xkvh,
                   const _Float16* wqT, const float* bq,
                   const _Float16* wkT, const float* bk,
                   const _Float16* wvT, const float* bv,
                   const _Float16* woT, const float* bo,
                   const int* padp, int causal) {
        gemm(xqh,  wqT, bq, nullptr, qh,  T, D, D, D, D, 0, D,
             0,0,0,0,0,0,0,0, 1, 1, 1.0f, 0, 0);
        gemm(xkvh, wkT, bk, nullptr, kh,  T, D, D, D, D, 0, D,
             0,0,0,0,0,0,0,0, 1, 1, 1.0f, 0, 0);
        gemm(xkvh, wvT, bv, nullptr, vT,  T, D, D, D, D, 0, 0,
             0,0,0,0,0,0,0,0, 1, 1, 1.0f, 0, 1);          // per-head transposed V
        gemm(qh, kh, nullptr, sc, nullptr, S, S, DK, D, D, S, 0,
             SD, DK, SD, DK, (long long)H * SS, SS, 0, 0,
             H, B * H, iskd, 0, 0);
        softmax_mask<<<dim3(S, B * H), TB, 0, stream>>>(sc, sch, padp, S, H, causal);
        gemm(sch, vT, nullptr, nullptr, t0h, S, DK, S, S, S, 0, D,
             (long long)H * SS, SS, (long long)H * DKS, DKS,
             0, 0, SD, DK, H, B * H, 1.0f, 0, 0);
        gemm(t0h, woT, bo, t1, nullptr, T, D, D, D, D, D, 0,
             0,0,0,0,0,0,0,0, 1, 1, 1.0f, 0, 0);
    };

    // ================= encoder =================
    embed_pe<<<T, TB, 0, stream>>>(enc_x, emb, x, xh, S, D);
    for (int l = 0; l < 6; ++l) {
        const size_t oDD = (size_t)l * DD, oD = (size_t)l * D;
        const size_t oDF = (size_t)l * DF, oF = (size_t)l * F;
        mha(xh, xh, eW[0]+oDD, P(5)+oD, eW[1]+oDD, P(7)+oD,
            eW[2]+oDD, P(9)+oD, eW[3]+oDD, P(11)+oD, pad, 0);
        add_layernorm<<<T, TB, 0, stream>>>(x, t1, P(12)+oD, P(13)+oD, x, xh, D);
        gemm(xh,  eW1+oDF, P(15)+oF, nullptr, ffh, T, F, D, D, D, 0, F,
             0,0,0,0,0,0,0,0, 1, 1, 1.0f, 1, 0);
        gemm(ffh, eW2+oDF, P(17)+oD, t1, nullptr, T, D, F, F, F, D, 0,
             0,0,0,0,0,0,0,0, 1, 1, 1.0f, 0, 0);
        add_layernorm<<<T, TB, 0, stream>>>(x, t1, P(18)+oD, P(19)+oD, x, xh, D);
    }
    hipMemcpyAsync(memh, xh, TD * sizeof(_Float16), hipMemcpyDeviceToDevice, stream);
    hipMemcpyAsync((float*)d_out + (size_t)T * V, x, TD * sizeof(float),
                   hipMemcpyDeviceToDevice, stream);

    // ================= decoder =================
    embed_pe<<<T, TB, 0, stream>>>(dec_x, emb, x, xh, S, D);
    for (int l = 0; l < 6; ++l) {
        const size_t oDD = (size_t)l * DD, oD = (size_t)l * D;
        const size_t oDF = (size_t)l * DF, oF = (size_t)l * F;
        mha(xh, xh, dW[0]+oDD, P(21)+oD, dW[1]+oDD, P(23)+oD,
            dW[2]+oDD, P(25)+oD, dW[3]+oDD, P(27)+oD, nullptr, 1);
        add_layernorm<<<T, TB, 0, stream>>>(x, t1, P(36)+oD, P(37)+oD, x, xh, D);
        mha(xh, memh, dW[4]+oDD, P(29)+oD, dW[5]+oDD, P(31)+oD,
            dW[6]+oDD, P(33)+oD, dW[7]+oDD, P(35)+oD, pad, 0);
        add_layernorm<<<T, TB, 0, stream>>>(x, t1, P(38)+oD, P(39)+oD, x, xh, D);
        gemm(xh,  dW1+oDF, P(43)+oF, nullptr, ffh, T, F, D, D, D, 0, F,
             0,0,0,0,0,0,0,0, 1, 1, 1.0f, 1, 0);
        gemm(ffh, dW2+oDF, P(45)+oD, t1, nullptr, T, D, F, F, F, D, 0,
             0,0,0,0,0,0,0,0, 1, 1, 1.0f, 0, 0);
        add_layernorm<<<T, TB, 0, stream>>>(x, t1, P(40)+oD, P(41)+oD, x, xh, D);
    }

    // tied logits = x @ emb^T  (Bt = embh [V,D])
    gemm(xh, embh, nullptr, (float*)d_out, nullptr, T, V, D, D, D, V, 0,
         0,0,0,0,0,0,0,0, 1, 1, 1.0f, 0, 0);
    #undef P
}
